// ModelGNN_35304631174019
// MI455X (gfx1250) — matrix operand order repaired and checked by reading, not compiled, not run on hardware
//
#include <hip/hip_runtime.h>
#include <hip/hip_bf16.h>

typedef __bf16 bf16_t;
typedef __attribute__((ext_vector_type(16))) __bf16 v16bf;
typedef __attribute__((ext_vector_type(8)))  float  v8f;

// Model dims (fixed by reference): F=7, H=300, L=100, G=64
#define HDIM 300
#define HPAD 304     // H padded to 19 N-tiles of 16
#define KHID 320     // H padded to multiple of 32 (K dimension)
#define LDIM 100
#define LPAD 112     // L padded to 7 N-tiles of 16
#define NT_H 19
#define NT_L 7
#define TM   32      // edges per workgroup (2 M-tiles of 16)

// ---------- order-preserving float <-> uint key (for atomic max) ----------
__device__ __forceinline__ unsigned fkey(float f) {
  unsigned u = __float_as_uint(f);
  return (u & 0x80000000u) ? ~u : (u | 0x80000000u);
}
// decode; clamps negatives and "empty" (key==0) to 0  == relu(where(cnt>0,max,0))
__device__ __forceinline__ float funkey(unsigned k) {
  return (k > 0x80000000u) ? __uint_as_float(k ^ 0x80000000u) : 0.0f;
}

union FragU { v16bf v; uint4 q[2]; };

// A fragment 16x32 bf16 from row-major LDS [rows][ldk].
// Lane l (<16): row M=l, K = {0..7, 16..23}; lane>=16: row M=l-16, K = {8..15, 24..31}
__device__ __forceinline__ v16bf load_a_frag(const bf16_t* lds, int row0, int ldk,
                                             int k0, int lane) {
  const int m = lane & 15, half = lane >> 4;
  const bf16_t* base = lds + (row0 + m) * ldk + k0 + half * 8;
  FragU u;
  u.q[0] = *(const uint4*)(base);        // K = k0+half*8 .. +7
  u.q[1] = *(const uint4*)(base + 16);   // K = k0+16+half*8 .. +7
  return u.v;
}

// B fragment 32x16 bf16 from transposed weights WT [Npad][Kpad] (element (k,n) at n*ldk+k).
// Lane l: col N = l%16, half=l/16: e<8 -> K = k0 + half*8 + e; e>=8 -> K = k0 + 16 + half*8 + (e-8)
__device__ __forceinline__ v16bf load_b_frag(const bf16_t* wt, int col0, int ldk,
                                             int k0, int lane) {
  const int n = lane & 15, half = lane >> 4;
  const bf16_t* base = wt + (size_t)(col0 + n) * ldk + k0 + half * 8;
  FragU u;
  u.q[0] = *(const uint4*)(base);
  u.q[1] = *(const uint4*)(base + 16);
  return u.v;
}

#define WMMA_BF16(C, A, B) \
  __builtin_amdgcn_wmma_f32_16x16x32_bf16(false, (A), false, (B), (short)0, (C), false, false)

// ---------- weight conversion: f32 [K][N] row-major -> bf16 [NP][KP] transposed ----------
__global__ void convert_wt_kernel(const float* __restrict__ w, bf16_t* __restrict__ wt,
                                  int K, int Nc, int KP, int NP) {
  int i = blockIdx.x * blockDim.x + threadIdx.x;
  if (i >= KP * NP) return;
  int n = i / KP, k = i % KP;
  float v = (k < K && n < Nc) ? w[k * Nc + n] : 0.0f;
  wt[i] = (bf16_t)v;
}

__global__ void zero_u32_kernel(unsigned* __restrict__ p, int n) {
  int i = blockIdx.x * blockDim.x + threadIdx.x;
  if (i < n) p[i] = 0u;
}

// ---------- fused edge layer: gather -> 3-layer MLP (WMMA bf16) -> scatter max ----------
template<int KP_IN, int LAYER>
__global__ __launch_bounds__(256) void edge_mlp_kernel(
    const float* __restrict__ x,          // LAYER==0: node features [N,7]
    const bf16_t* __restrict__ nodeh,     // LAYER==1: node features [N,LPAD] bf16
    const int* __restrict__ ei,           // [2,E]
    const bf16_t* __restrict__ w1t, const float* __restrict__ b1,
    const bf16_t* __restrict__ w2t, const float* __restrict__ b2,
    const bf16_t* __restrict__ w3t, const float* __restrict__ b3,
    unsigned* __restrict__ keys,          // [N,LDIM] keyed max accumulator
    int E) {
  __shared__ __align__(16) bf16_t sA [TM * KP_IN];
  __shared__ __align__(16) bf16_t sH1[TM * KHID];
  __shared__ __align__(16) bf16_t sH2[TM * KHID];
  __shared__ float sB1[HPAD];
  __shared__ float sB2[HPAD];
  __shared__ float sB3[LPAD];
  __shared__ int sSrc[TM];
  __shared__ int sDst[TM];

  const int tid = threadIdx.x;
  const int e0  = blockIdx.x * TM;

  if (tid < TM) {
    int e = e0 + tid;
    sSrc[tid] = (e < E) ? ei[e] : 0;
    sDst[tid] = (e < E) ? ei[E + e] : 0;
  }
  // stage zero-padded biases in LDS (removes guarded global loads from tile loops)
  for (int i = tid; i < HPAD; i += 256) {
    sB1[i] = (i < HDIM) ? b1[i] : 0.0f;
    sB2[i] = (i < HDIM) ? b2[i] : 0.0f;
  }
  for (int i = tid; i < LPAD; i += 256) sB3[i] = (i < LDIM) ? b3[i] : 0.0f;
  // zero the K-pad columns (304..319) of the hidden buffers
  for (int i = tid; i < TM * 16; i += 256) {
    int m = i >> 4, c = HPAD + (i & 15);
    sH1[m * KHID + c] = (bf16_t)0.0f;
    sH2[m * KHID + c] = (bf16_t)0.0f;
  }
  __syncthreads();

  // build per-edge input [x_i, x_j - x_i] (bf16, zero-padded to KP_IN)
  for (int i = tid; i < TM * KP_IN; i += 256) {
    int m = i / KP_IN, c = i % KP_IN;
    float v = 0.0f;
    if (LAYER == 0) {
      if (c < 7)       v = x[sDst[m] * 7 + c];
      else if (c < 14) { int cc = c - 7; v = x[sSrc[m]*7 + cc] - x[sDst[m]*7 + cc]; }
    } else {
      if (c < LDIM)        v = (float)nodeh[(size_t)sDst[m] * LPAD + c];
      else if (c < 2*LDIM) { int cc = c - LDIM;
        v = (float)nodeh[(size_t)sSrc[m]*LPAD + cc] - (float)nodeh[(size_t)sDst[m]*LPAD + cc]; }
    }
    sA[i] = (bf16_t)v;
  }
  __syncthreads();

  const int wave = tid >> 5;
  const int lane = tid & 31;
  const int n16  = lane & 15;
  const int mh   = (lane >> 4) * 8;   // C/D: lane l, vgpr v -> M = v + 8*(l/16), N = l%16

  // GEMM1: relu(sA @ W1 + b1) -> sH1
  for (int nt = wave; nt < NT_H; nt += 8) {
    v8f c0 = {}; v8f c1 = {};
#pragma unroll 2
    for (int k = 0; k < KP_IN; k += 32) {
      v16bf b  = load_b_frag(w1t, nt * 16, KP_IN, k, lane);
      v16bf a0 = load_a_frag(sA, 0,  KP_IN, k, lane);
      v16bf a1 = load_a_frag(sA, 16, KP_IN, k, lane);
      c0 = WMMA_BF16(c0, a0, b);
      c1 = WMMA_BF16(c1, a1, b);
    }
    const int col = nt * 16 + n16;
    const float bias = sB1[col];
    for (int v = 0; v < 8; v++) {
      sH1[(mh + v)      * KHID + col] = (bf16_t)fmaxf(c0[v] + bias, 0.0f);
      sH1[(16 + mh + v) * KHID + col] = (bf16_t)fmaxf(c1[v] + bias, 0.0f);
    }
  }
  __syncthreads();

  // GEMM2: relu(sH1 @ W2 + b2) -> sH2
  for (int nt = wave; nt < NT_H; nt += 8) {
    v8f c0 = {}; v8f c1 = {};
#pragma unroll 2
    for (int k = 0; k < KHID; k += 32) {
      v16bf b  = load_b_frag(w2t, nt * 16, KHID, k, lane);
      v16bf a0 = load_a_frag(sH1, 0,  KHID, k, lane);
      v16bf a1 = load_a_frag(sH1, 16, KHID, k, lane);
      c0 = WMMA_BF16(c0, a0, b);
      c1 = WMMA_BF16(c1, a1, b);
    }
    const int col = nt * 16 + n16;
    const float bias = sB2[col];
    for (int v = 0; v < 8; v++) {
      sH2[(mh + v)      * KHID + col] = (bf16_t)fmaxf(c0[v] + bias, 0.0f);
      sH2[(16 + mh + v) * KHID + col] = (bf16_t)fmaxf(c1[v] + bias, 0.0f);
    }
  }
  __syncthreads();

  // GEMM3: msg = sH2 @ W3 + b3, fused scatter-max(dst) via keyed atomicMax
  for (int nt = wave; nt < NT_L; nt += 8) {
    v8f c0 = {}; v8f c1 = {};
#pragma unroll 2
    for (int k = 0; k < KHID; k += 32) {
      v16bf b  = load_b_frag(w3t, nt * 16, KHID, k, lane);
      v16bf a0 = load_a_frag(sH2, 0,  KHID, k, lane);
      v16bf a1 = load_a_frag(sH2, 16, KHID, k, lane);
      c0 = WMMA_BF16(c0, a0, b);
      c1 = WMMA_BF16(c1, a1, b);
    }
    const int col = nt * 16 + n16;
    if (col < LDIM) {
      const float bias = sB3[col];
      for (int v = 0; v < 8; v++) {
        const int m0 = mh + v, m1 = 16 + mh + v;
        if (e0 + m0 < E)
          atomicMax(&keys[(size_t)sDst[m0] * LDIM + col], fkey(c0[v] + bias));
        if (e0 + m1 < E)
          atomicMax(&keys[(size_t)sDst[m1] * LDIM + col], fkey(c1[v] + bias));
      }
    }
  }
}

// decode keyed max -> bf16 node features [N,LPAD] (pad cols zero) and re-arm keys
__global__ void decode_bf16_kernel(unsigned* __restrict__ keys, bf16_t* __restrict__ nh, int Nn) {
  int i = blockIdx.x * blockDim.x + threadIdx.x;
  if (i >= Nn * LPAD) return;
  int node = i / LPAD, c = i % LPAD;
  float v = 0.0f;
  if (c < LDIM) {
    int ki = node * LDIM + c;
    v = funkey(keys[ki]);
    keys[ki] = 0u;
  }
  nh[i] = (bf16_t)v;
}

// decode keyed max -> f32 node features [N,LDIM]
__global__ void decode_f32_kernel(const unsigned* __restrict__ keys, float* __restrict__ out,
                                  int total) {
  int i = blockIdx.x * blockDim.x + threadIdx.x;
  if (i >= total) return;
  out[i] = funkey(keys[i]);
}

// pooling over sorted batch: run-length accumulate then flush with atomics
__global__ __launch_bounds__(128) void pool_kernel(const float* __restrict__ h2,
                                                   const int* __restrict__ batch,
                                                   float* __restrict__ addp,
                                                   unsigned* __restrict__ maxk,
                                                   float* __restrict__ cnt, int Nn) {
  __shared__ int sb[128];
  const int chunk = blockIdx.x * 128;
  const int len = (Nn - chunk < 128) ? (Nn - chunk) : 128;
  if ((int)threadIdx.x < len) sb[threadIdx.x] = batch[chunk + threadIdx.x];
  __syncthreads();
  const int c = threadIdx.x;
  if (c < LDIM) {
    int curg = -1; float s = 0.0f, mx = 0.0f;   // h2 >= 0 so mx=0 init is safe
    for (int i = 0; i < len; i++) {
      int g = sb[i];
      if (g != curg) {
        if (curg >= 0) { atomicAdd(&addp[curg * LDIM + c], s);
                         atomicMax(&maxk[curg * LDIM + c], fkey(mx)); }
        curg = g; s = 0.0f; mx = 0.0f;
      }
      float v = h2[(size_t)(chunk + i) * LDIM + c];
      s += v; mx = fmaxf(mx, v);
    }
    if (curg >= 0) { atomicAdd(&addp[curg * LDIM + c], s);
                     atomicMax(&maxk[curg * LDIM + c], fkey(mx)); }
  } else if (c == LDIM) {
    int curg = -1; float s = 0.0f;
    for (int i = 0; i < len; i++) {
      int g = sb[i];
      if (g != curg) { if (curg >= 0) atomicAdd(&cnt[curg], s); curg = g; s = 0.0f; }
      s += 1.0f;
    }
    if (curg >= 0) atomicAdd(&cnt[curg], s);
  }
}

// pooled[g] = [addp, addp/max(cnt,1), max (0 if empty), u]  -> [G,302]
__global__ void build_pooled_kernel(const float* __restrict__ addp,
                                    const unsigned* __restrict__ maxk,
                                    const float* __restrict__ cnt,
                                    const float* __restrict__ u,
                                    float* __restrict__ pooled, int G) {
  int i = blockIdx.x * blockDim.x + threadIdx.x;
  if (i >= G * 302) return;
  int g = i / 302, c = i % 302;
  float v;
  if (c < 100)       v = addp[g * 100 + c];
  else if (c < 200)  v = addp[g * 100 + (c - 100)] / fmaxf(cnt[g], 1.0f);
  else if (c < 300)  v = funkey(maxk[g * 100 + (c - 200)]);
  else               v = u[g * 2 + (c - 300)];
  pooled[i] = v;
}

// tiny dense layer: out[g][o] = act(sum_k in[g][k] * w[k][o] + b[o])
__global__ void dense_kernel(const float* __restrict__ in, const float* __restrict__ w,
                             const float* __restrict__ b, float* __restrict__ out,
                             int K, int O, int relu, int G) {
  int i = blockIdx.x * blockDim.x + threadIdx.x;
  if (i >= G * O) return;
  int g = i / O, o = i % O;
  float s = b[o];
  for (int k = 0; k < K; k++) s += in[g * K + k] * w[k * O + o];
  if (relu) s = fmaxf(s, 0.0f);
  out[i] = s;
}

static inline int cdiv(long a, long b) { return (int)((a + b - 1) / b); }

extern "C" void kernel_launch(void* const* d_in, const int* in_sizes, int n_in,
                              void* d_out, int out_size, void* d_ws, size_t ws_size,
                              hipStream_t stream) {
  const float* x     = (const float*)d_in[0];
  const int*   ei    = (const int*)d_in[1];
  const int*   batch = (const int*)d_in[2];
  const float* u     = (const float*)d_in[3];
  const float* l0_w1 = (const float*)d_in[4];  const float* l0_b1 = (const float*)d_in[5];
  const float* l0_w2 = (const float*)d_in[6];  const float* l0_b2 = (const float*)d_in[7];
  const float* l0_w3 = (const float*)d_in[8];  const float* l0_b3 = (const float*)d_in[9];
  const float* l1_w1 = (const float*)d_in[10]; const float* l1_b1 = (const float*)d_in[11];
  const float* l1_w2 = (const float*)d_in[12]; const float* l1_b2 = (const float*)d_in[13];
  const float* l1_w3 = (const float*)d_in[14]; const float* l1_b3 = (const float*)d_in[15];
  const float* lw1   = (const float*)d_in[16]; const float* lb1   = (const float*)d_in[17];
  const float* lw2   = (const float*)d_in[18]; const float* lb2   = (const float*)d_in[19];
  const float* lw3   = (const float*)d_in[20]; const float* lb3   = (const float*)d_in[21];
  (void)n_in; (void)out_size; (void)ws_size;

  const int Nn = in_sizes[0] / 7;
  const int E  = in_sizes[1] / 2;
  const int G  = in_sizes[3] / 2;

  char* wsb = (char*)d_ws;
  size_t off = 0;
  auto alloc = [&](size_t bytes) -> char* {
    char* p = wsb + off;
    off = (off + bytes + 255) & ~(size_t)255;
    return p;
  };
  unsigned* keys   = (unsigned*)alloc((size_t)Nn * LDIM * 4);
  bf16_t*   nh1    = (bf16_t*)  alloc((size_t)Nn * LPAD * 2);
  float*    nh2    = (float*)   alloc((size_t)Nn * LDIM * 4);
  float*    addp   = (float*)   alloc((size_t)G * LDIM * 4);
  unsigned* maxk   = (unsigned*)alloc((size_t)G * LDIM * 4);
  float*    cnt    = (float*)   alloc((size_t)G * 4);
  float*    pooled = (float*)   alloc((size_t)G * 302 * 4);
  float*    t1     = (float*)   alloc((size_t)G * LDIM * 4);
  float*    t2     = (float*)   alloc((size_t)G * LDIM * 4);
  bf16_t*   w01t   = (bf16_t*)  alloc((size_t)HPAD * 32   * 2);
  bf16_t*   w02t   = (bf16_t*)  alloc((size_t)HPAD * KHID * 2);
  bf16_t*   w03t   = (bf16_t*)  alloc((size_t)LPAD * KHID * 2);
  bf16_t*   w11t   = (bf16_t*)  alloc((size_t)HPAD * 224  * 2);
  bf16_t*   w12t   = (bf16_t*)  alloc((size_t)HPAD * KHID * 2);
  bf16_t*   w13t   = (bf16_t*)  alloc((size_t)LPAD * KHID * 2);

  // zero accumulators (d_ws is poisoned; re-zero every call for determinism)
  zero_u32_kernel<<<cdiv((long)Nn * LDIM, 256), 256, 0, stream>>>(keys, Nn * LDIM);
  zero_u32_kernel<<<cdiv((long)G * LDIM, 256), 256, 0, stream>>>((unsigned*)addp, G * LDIM);
  zero_u32_kernel<<<cdiv((long)G * LDIM, 256), 256, 0, stream>>>(maxk, G * LDIM);
  zero_u32_kernel<<<1, 256, 0, stream>>>((unsigned*)cnt, G);

  // bf16-transpose-pad all edge-MLP weights
  convert_wt_kernel<<<cdiv(HPAD * 32, 256),   256, 0, stream>>>(l0_w1, w01t, 14,  HDIM, 32,   HPAD);
  convert_wt_kernel<<<cdiv(HPAD * KHID, 256), 256, 0, stream>>>(l0_w2, w02t, HDIM, HDIM, KHID, HPAD);
  convert_wt_kernel<<<cdiv(LPAD * KHID, 256), 256, 0, stream>>>(l0_w3, w03t, HDIM, LDIM, KHID, LPAD);
  convert_wt_kernel<<<cdiv(HPAD * 224, 256),  256, 0, stream>>>(l1_w1, w11t, 200, HDIM, 224,  HPAD);
  convert_wt_kernel<<<cdiv(HPAD * KHID, 256), 256, 0, stream>>>(l1_w2, w12t, HDIM, HDIM, KHID, HPAD);
  convert_wt_kernel<<<cdiv(LPAD * KHID, 256), 256, 0, stream>>>(l1_w3, w13t, HDIM, LDIM, KHID, LPAD);

  // edge layer 0 (input F=7 -> K padded 32)
  edge_mlp_kernel<32, 0><<<cdiv(E, TM), 256, 0, stream>>>(
      x, nullptr, ei, w01t, l0_b1, w02t, l0_b2, w03t, l0_b3, keys, E);
  decode_bf16_kernel<<<cdiv((long)Nn * LPAD, 256), 256, 0, stream>>>(keys, nh1, Nn);

  // edge layer 1 (input 2L=200 -> K padded 224)
  edge_mlp_kernel<224, 1><<<cdiv(E, TM), 256, 0, stream>>>(
      nullptr, nh1, ei, w11t, l1_b1, w12t, l1_b2, w13t, l1_b3, keys, E);
  decode_f32_kernel<<<cdiv((long)Nn * LDIM, 256), 256, 0, stream>>>(keys, nh2, Nn * LDIM);

  // global add/mean/max pooling + final MLP
  pool_kernel<<<cdiv(Nn, 128), 128, 0, stream>>>(nh2, batch, addp, maxk, cnt, Nn);
  build_pooled_kernel<<<cdiv((long)G * 302, 256), 256, 0, stream>>>(addp, maxk, cnt, u, pooled, G);
  dense_kernel<<<cdiv((long)G * LDIM, 256), 256, 0, stream>>>(pooled, lw1, lb1, t1, 302, LDIM, 1, G);
  dense_kernel<<<cdiv((long)G * LDIM, 256), 256, 0, stream>>>(t1, lw2, lb2, t2, LDIM, LDIM, 1, G);
  dense_kernel<<<cdiv((long)G * 2, 256), 256, 0, stream>>>(t2, lw3, lb3, (float*)d_out, LDIM, 2, 0, G);
}
